// TransitionGNN_24713241822152
// MI455X (gfx1250) — matrix operand order repaired and checked by reading, not compiled, not run on hardware
//
#include <hip/hip_runtime.h>
#include <hip/hip_bf16.h>

// ---------------------------------------------------------------------------
// CDNA5 (gfx1250) fused TransitionGNN.
// bf16 WMMA 16x16x32, f32 accumulate. One workgroup per batch element for the
// edge MLP + segment-sum (all activations resident in the 320KB WGP LDS);
// weights converted once to bf16 in B-fragment-swizzled order so every WMMA
// B operand is one contiguous 32B load per lane. Bulk input staging uses the
// gfx1250 async global->LDS DMA path (ASYNCcnt) when the toolchain has it.
// ---------------------------------------------------------------------------

typedef __attribute__((ext_vector_type(16))) __bf16 v16bf;
typedef __attribute__((ext_vector_type(8)))  float  v8f;
typedef int v4i __attribute__((vector_size(16)));   // matches async b128 builtin param

#define WMMA_BF16(a, b, c) \
  __builtin_amdgcn_wmma_f32_16x16x32_bf16(false, (a), false, (b), (short)0, (c), false, false)

#define AS1 __attribute__((address_space(1)))
#define AS3 __attribute__((address_space(3)))

#if __has_builtin(__builtin_amdgcn_global_load_async_to_lds_b32) && \
    __has_builtin(__builtin_amdgcn_global_load_async_to_lds_b128) && \
    __has_builtin(__builtin_amdgcn_s_wait_asynccnt)
#define USE_ASYNC_LDS 1
#else
#define USE_ASYNC_LDS 0
#endif

// ---------------------------------------------------------------------------
// Weight prep: f32 row-major [Ksrc x N] -> bf16 tiles in B-fragment order.
// Tile (nt, kt) stored at (nt*Ktiles + kt)*512; within a tile, element
// [lane*16 + e] = W[kt*32 + 8*(lane>>4) + (e&7) + 16*(e>>3), nt*16 + (lane&15)].
// Rows >= Ksrc are zero padding (used for the 548->576 pad of nW1).
// ---------------------------------------------------------------------------
__global__ __launch_bounds__(256) void swizzle_w_kernel(
    const float* __restrict__ src, __bf16* __restrict__ dst,
    int Ksrc, int N, int Ktiles, int total) {
  int idx = blockIdx.x * 256 + threadIdx.x;
  if (idx >= total) return;
  int tile = idx >> 9;
  int r    = idx & 511;
  int lane = r >> 4, e = r & 15;
  int nt = tile / Ktiles;
  int kt = tile - nt * Ktiles;
  int k  = (kt << 5) + ((lane >> 4) << 3) + (e & 7) + ((e >> 3) << 4);
  int n  = (nt << 4) + (lane & 15);
  float v = (k < Ksrc) ? src[(size_t)k * N + n] : 0.f;
  dst[idx] = (__bf16)v;
}

// ---------------------------------------------------------------------------
// Kernel 1: edge MLP + LayerNorm + segment-sum.  grid = B = 2048 blocks,
// block = 256 threads (8 wave32s).  Rows = 96 (90 edges + 6 zero-pad rows).
//
// LDS layout (dynamic, 232192 B total):
//   sA     : edge_in A-fragments,  6 Mtiles x  2 Ktiles x 512 bf16 =  12288 B
//   sH1    : h1  A-fragments,      6 Mtiles x 16 Ktiles x 512 bf16 =  98304 B
//   sH2    : h2  A-fragments,      same                            =  98304 B
//   sAgg   : per-node agg,         10 x 512 f32                    =  20480 B
//   stats  : sum/sumsq/mean/inv,   4 x 96 f32                      =   1536 B
//   sStates: staged node block,    10 x 32 f32                     =   1280 B
// ---------------------------------------------------------------------------
__global__ __launch_bounds__(256) void edge_mlp_agg_kernel(
    const float* __restrict__ states,
    const __bf16* __restrict__ W1, const float* __restrict__ b1,
    const __bf16* __restrict__ W2, const float* __restrict__ b2,
    const float* __restrict__ gam, const float* __restrict__ bet,
    const __bf16* __restrict__ W3, const float* __restrict__ b3,
    __bf16* __restrict__ agg_out) {
  extern __shared__ char smem[];
  __bf16* sA      = (__bf16*)smem;                           // 6*2*512
  __bf16* sH1     = (__bf16*)(smem + 12288);                 // 6*16*512
  __bf16* sH2     = (__bf16*)(smem + 12288 + 98304);         // 6*16*512
  float*  sAgg    = (float*)(smem + 12288 + 2 * 98304);      // 10*512
  float*  sSum    = (float*)(smem + 12288 + 2 * 98304 + 20480);
  float*  sSq     = sSum + 96;
  float*  sMean   = sSq + 96;
  float*  sInv    = sMean + 96;
  float*  sStates = (float*)(smem + 230912);                 // 10*32 f32

  const int tid   = threadIdx.x;
  const int lane  = tid & 31;
  const int wave  = tid >> 5;
  const int khalf = lane >> 4;
  const int batch = blockIdx.x;

  // ---- Phase 0a: async-stage this batch's 10x32 node block into LDS,
  //      overlap with accumulator zeroing.
#if USE_ASYNC_LDS
  for (int i = tid; i < 320; i += 256)
    __builtin_amdgcn_global_load_async_to_lds_b32(
        (AS1 int*)(void*)(states + (size_t)batch * 320 + i),
        (AS3 int*)(void*)(sStates + i), 0, 0);
#else
  for (int i = tid; i < 320; i += 256)
    sStates[i] = states[(size_t)batch * 320 + i];
#endif
  for (int i = tid; i < 5120; i += 256) sAgg[i] = 0.f;
  if (tid < 96) { sSum[tid] = 0.f; sSq[tid] = 0.f; }
#if USE_ASYNC_LDS
  __builtin_amdgcn_s_wait_asynccnt(0);
#endif
  __syncthreads();

  // ---- Phase 0b: build edge_in [96 x 64] in A-frag order from staged nodes
  for (int idx = tid; idx < 96 * 64; idx += 256) {
    int e = idx >> 6;      // edge row (0..95; >=90 is zero pad)
    int c = idx & 63;      // input column (0..63)
    float v = 0.f;
    if (e < 90) {
      int i  = e / 9;
      int jj = e - i * 9;
      int j  = jj + (jj >= i);          // skip self-loop
      int node = (c < 32) ? i : j;
      v = sStates[node * 32 + (c & 31)];
    }
    int mt = e >> 4, m = e & 15;
    int kt = c >> 5, kin = c & 31;
    int l2 = m + (((kin >> 3) & 1) << 4);
    int e2 = (kin & 7) | ((kin >> 4) << 3);
    sA[(((mt << 1) + kt) << 9) + (l2 << 4) + e2] = (__bf16)v;
  }
  __syncthreads();

  // ---- Phase 1: h1 = relu(edge_in @ W1 + b1)  (K=64, 2 wmma steps/tile)
  for (int t = wave; t < 6 * 32; t += 8) {
    int mt = t >> 5, nt = t & 31;
    v8f acc = {};
    for (int kt = 0; kt < 2; ++kt) {
      v16bf a  = ((const v16bf*)(sA + (((mt << 1) + kt) << 9)))[lane];
      v16bf bb = ((const v16bf*)(W1 + (((nt << 1) + kt) << 9)))[lane];
      acc = WMMA_BF16(a, bb, acc);
    }
    int n    = (nt << 4) + (lane & 15);
    float bv = b1[n];
    int kt2  = n >> 5, kin = n & 31;
    int l2b  = ((kin >> 3) & 1) << 4;
    int e2   = (kin & 7) | ((kin >> 4) << 3);
    for (int r = 0; r < 8; ++r) {
      float v = acc[r] + bv;
      v = v > 0.f ? v : 0.f;
      int m2 = r + (khalf << 3);
      sH1[(((mt << 4) + kt2) << 9) + ((m2 + l2b) << 4) + e2] = (__bf16)v;
    }
  }
  __syncthreads();

  // ---- Phase 2: h2 = h1 @ W2 + b2, LN stats via LDS float atomics
  for (int t = wave; t < 6 * 32; t += 8) {
    int mt = t >> 5, nt = t & 31;
    v8f acc = {};
    for (int kt = 0; kt < 16; ++kt) {
      if (kt + 1 < 16) __builtin_prefetch(W2 + ((((nt << 4) + kt + 1) << 9)), 0, 0);
      v16bf a  = ((const v16bf*)(sH1 + (((mt << 4) + kt) << 9)))[lane];
      v16bf bb = ((const v16bf*)(W2  + (((nt << 4) + kt) << 9)))[lane];
      acc = WMMA_BF16(a, bb, acc);
    }
    int n    = (nt << 4) + (lane & 15);
    float bv = b2[n];
    int kt2  = n >> 5, kin = n & 31;
    int l2b  = ((kin >> 3) & 1) << 4;
    int e2   = (kin & 7) | ((kin >> 4) << 3);
    for (int r = 0; r < 8; ++r) {
      int m   = (mt << 4) + r + (khalf << 3);
      float v = acc[r] + bv;
      atomicAdd(&sSum[m], v);
      atomicAdd(&sSq[m], v * v);
      sH2[(((mt << 4) + kt2) << 9) + (((r + (khalf << 3)) + l2b) << 4) + e2] = (__bf16)v;
    }
  }
  __syncthreads();
  if (tid < 96) {
    float mean = sSum[tid] * (1.f / 512.f);
    float var  = sSq[tid] * (1.f / 512.f) - mean * mean;
    sMean[tid] = mean;
    sInv[tid]  = rsqrtf(var + 1e-5f);
  }
  __syncthreads();
  // normalize + affine + relu in place (bf16 in A-frag order)
  for (int idx = tid; idx < 6 * 16 * 512; idx += 256) {
    int tile = idx >> 9;
    int mt = tile >> 4, kt = tile & 15;
    int r = idx & 511;
    int l = r >> 4, e = r & 15;
    int m = (mt << 4) + (l & 15);
    int k = (kt << 5) + (e & 7) + ((((e >> 3) << 1) + (l >> 4)) << 3);
    float v = (float)sH2[idx];
    v = (v - sMean[m]) * sInv[m] * gam[k] + bet[k];
    v = v > 0.f ? v : 0.f;
    sH2[idx] = (__bf16)v;
  }
  __syncthreads();

  // ---- Phase 3: edge_attr = h2 @ W3 + b3, fused segment-sum into sAgg
  for (int t = wave; t < 6 * 32; t += 8) {
    int mt = t >> 5, nt = t & 31;
    v8f acc = {};
    for (int kt = 0; kt < 16; ++kt) {
      v16bf a  = ((const v16bf*)(sH2 + (((mt << 4) + kt) << 9)))[lane];
      v16bf bb = ((const v16bf*)(W3  + (((nt << 4) + kt) << 9)))[lane];
      acc = WMMA_BF16(a, bb, acc);
    }
    int n    = (nt << 4) + (lane & 15);
    float bv = b3[n];
    for (int r = 0; r < 8; ++r) {
      int m = (mt << 4) + r + (khalf << 3);
      if (m < 90) {
        int node = m / 9;   // edges are row-major: 9 consecutive edges per src node
        atomicAdd(&sAgg[(node << 9) + n], acc[r] + bv);
      }
    }
  }
  __syncthreads();
  for (int i = tid; i < 5120; i += 256)
    agg_out[(size_t)(batch * 10) * 512 + i] = (__bf16)sAgg[i];
}

// ---------------------------------------------------------------------------
// Kernel 2: node MLP.  64 rows per block -> 320 blocks, 8 waves.
// node_in = [nodes(32) | action(4) | agg(512)] -> K padded to 576 (18 ktiles).
// LDS: sA 73728 + sH1 65536 + sH2 65536 + stats 1024 + aggStage 65536
//    = 271360 B.
// ---------------------------------------------------------------------------
__global__ __launch_bounds__(256) void node_mlp_kernel(
    const float* __restrict__ states, const float* __restrict__ action,
    const __bf16* __restrict__ agg,
    const __bf16* __restrict__ W1, const float* __restrict__ b1,
    const __bf16* __restrict__ W2, const float* __restrict__ b2,
    const float* __restrict__ gam, const float* __restrict__ bet,
    const __bf16* __restrict__ W3, const float* __restrict__ b3,
    float* __restrict__ out) {
  extern __shared__ char smem[];
  __bf16* sA    = (__bf16*)smem;                       // 4*18*512
  __bf16* sH1   = (__bf16*)(smem + 73728);             // 4*16*512
  __bf16* sH2   = (__bf16*)(smem + 73728 + 65536);     // 4*16*512
  float*  sSum  = (float*)(smem + 73728 + 2 * 65536);
  float*  sSq   = sSum + 64;
  float*  sMean = sSq + 64;
  float*  sInv  = sMean + 64;
  __bf16* sAgS  = (__bf16*)(smem + 73728 + 2 * 65536 + 1024);  // 64*512 bf16

  const int tid = threadIdx.x, lane = tid & 31, wave = tid >> 5, khalf = lane >> 4;
  const int row0 = blockIdx.x * 64;

  // ---- Phase 0a: async-stage 64 contiguous agg rows (64KB bf16) into LDS
#if USE_ASYNC_LDS
  {
    const char* gsrc = (const char*)(agg + (size_t)row0 * 512);
    char* ldst = (char*)sAgS;
    for (int i = tid; i < 4096; i += 256)
      __builtin_amdgcn_global_load_async_to_lds_b128(
          (AS1 v4i*)(void*)(gsrc + (size_t)i * 16),
          (AS3 v4i*)(void*)(ldst + (size_t)i * 16), 0, 0);
  }
#else
  {
    const int4* gsrc = (const int4*)(agg + (size_t)row0 * 512);
    int4* ldst = (int4*)sAgS;
    for (int i = tid; i < 4096; i += 256) ldst[i] = gsrc[i];
  }
#endif
  if (tid < 64) { sSum[tid] = 0.f; sSq[tid] = 0.f; }
#if USE_ASYNC_LDS
  __builtin_amdgcn_s_wait_asynccnt(0);
#endif
  __syncthreads();

  // ---- Phase 0b: assemble node_in [64 x 576] in A-frag order
  for (int idx = tid; idx < 64 * 576; idx += 256) {
    int m = idx / 576;
    int c = idx - m * 576;
    int row = row0 + m;
    float v = 0.f;
    if (c < 32)       v = states[(size_t)row * 32 + c];
    else if (c < 36)  v = action[(row / 10) * 4 + (c - 32)];
    else if (c < 548) v = (float)sAgS[m * 512 + (c - 36)];
    int mt = m >> 4, mm = m & 15;
    int kt = c >> 5, kin = c & 31;
    int l2 = mm + (((kin >> 3) & 1) << 4);
    int e2 = (kin & 7) | ((kin >> 4) << 3);
    sA[((mt * 18 + kt) << 9) + (l2 << 4) + e2] = (__bf16)v;
  }
  __syncthreads();

  // ---- Layer 1: h1 = relu(node_in @ W1 + b1)  (18 wmma steps/tile)
  for (int t = wave; t < 4 * 32; t += 8) {
    int mt = t >> 5, nt = t & 31;
    v8f acc = {};
    for (int kt = 0; kt < 18; ++kt) {
      if (kt + 1 < 18) __builtin_prefetch(W1 + ((nt * 18 + kt + 1) << 9), 0, 0);
      v16bf a  = ((const v16bf*)(sA + ((mt * 18 + kt) << 9)))[lane];
      v16bf bb = ((const v16bf*)(W1 + ((nt * 18 + kt) << 9)))[lane];
      acc = WMMA_BF16(a, bb, acc);
    }
    int n    = (nt << 4) + (lane & 15);
    float bv = b1[n];
    int kt2  = n >> 5, kin = n & 31;
    int l2b  = ((kin >> 3) & 1) << 4;
    int e2   = (kin & 7) | ((kin >> 4) << 3);
    for (int r = 0; r < 8; ++r) {
      float v = acc[r] + bv;
      v = v > 0.f ? v : 0.f;
      int m2 = r + (khalf << 3);
      sH1[(((mt << 4) + kt2) << 9) + ((m2 + l2b) << 4) + e2] = (__bf16)v;
    }
  }
  __syncthreads();

  // ---- Layer 2: h2 = h1 @ W2 + b2, LN stats
  for (int t = wave; t < 4 * 32; t += 8) {
    int mt = t >> 5, nt = t & 31;
    v8f acc = {};
    for (int kt = 0; kt < 16; ++kt) {
      v16bf a  = ((const v16bf*)(sH1 + (((mt << 4) + kt) << 9)))[lane];
      v16bf bb = ((const v16bf*)(W2  + (((nt << 4) + kt) << 9)))[lane];
      acc = WMMA_BF16(a, bb, acc);
    }
    int n    = (nt << 4) + (lane & 15);
    float bv = b2[n];
    int kt2  = n >> 5, kin = n & 31;
    int l2b  = ((kin >> 3) & 1) << 4;
    int e2   = (kin & 7) | ((kin >> 4) << 3);
    for (int r = 0; r < 8; ++r) {
      int m   = (mt << 4) + r + (khalf << 3);
      float v = acc[r] + bv;
      atomicAdd(&sSum[m], v);
      atomicAdd(&sSq[m], v * v);
      sH2[(((mt << 4) + kt2) << 9) + (((r + (khalf << 3)) + l2b) << 4) + e2] = (__bf16)v;
    }
  }
  __syncthreads();
  if (tid < 64) {
    float mean = sSum[tid] * (1.f / 512.f);
    float var  = sSq[tid] * (1.f / 512.f) - mean * mean;
    sMean[tid] = mean;
    sInv[tid]  = rsqrtf(var + 1e-5f);
  }
  __syncthreads();
  for (int idx = tid; idx < 4 * 16 * 512; idx += 256) {
    int tile = idx >> 9;
    int mt = tile >> 4, kt = tile & 15;
    int r = idx & 511;
    int l = r >> 4, e = r & 15;
    int m = (mt << 4) + (l & 15);
    int k = (kt << 5) + (e & 7) + ((((e >> 3) << 1) + (l >> 4)) << 3);
    float v = (float)sH2[idx];
    v = (v - sMean[m]) * sInv[m] * gam[k] + bet[k];
    v = v > 0.f ? v : 0.f;
    sH2[idx] = (__bf16)v;
  }
  __syncthreads();

  // ---- Layer 3: out = h2 @ W3 + b3  (N=32 -> 2 ntiles; 8 tiles total)
  for (int t = wave; t < 8; t += 8) {
    int mt = t >> 1, nt = t & 1;
    v8f acc = {};
    for (int kt = 0; kt < 16; ++kt) {
      v16bf a  = ((const v16bf*)(sH2 + (((mt << 4) + kt) << 9)))[lane];
      v16bf bb = ((const v16bf*)(W3  + (((nt << 4) + kt) << 9)))[lane];
      acc = WMMA_BF16(a, bb, acc);
    }
    int n    = (nt << 4) + (lane & 15);
    float bv = b3[n];
    for (int r = 0; r < 8; ++r) {
      int m = (mt << 4) + r + (khalf << 3);
      out[(size_t)(row0 + m) * 32 + n] = acc[r] + bv;
    }
  }
}

// ---------------------------------------------------------------------------
// Host launcher
// ---------------------------------------------------------------------------
extern "C" void kernel_launch(void* const* d_in, const int* in_sizes, int n_in,
                              void* d_out, int out_size, void* d_ws, size_t ws_size,
                              hipStream_t stream) {
  (void)in_sizes; (void)n_in; (void)out_size; (void)ws_size;
  const float* states = (const float*)d_in[0];
  const float* action = (const float*)d_in[1];
  const float* eW1 = (const float*)d_in[2];  const float* eb1 = (const float*)d_in[3];
  const float* eW2 = (const float*)d_in[4];  const float* eb2 = (const float*)d_in[5];
  const float* eg  = (const float*)d_in[6];  const float* ebt = (const float*)d_in[7];
  const float* eW3 = (const float*)d_in[8];  const float* eb3 = (const float*)d_in[9];
  const float* nW1 = (const float*)d_in[10]; const float* nb1 = (const float*)d_in[11];
  const float* nW2 = (const float*)d_in[12]; const float* nb2 = (const float*)d_in[13];
  const float* ng  = (const float*)d_in[14]; const float* nbt = (const float*)d_in[15];
  const float* nW3 = (const float*)d_in[16]; const float* nb3 = (const float*)d_in[17];

  char* ws = (char*)d_ws;
  size_t off = 0;
  // swizzled bf16 weights in workspace
  __bf16* eW1s = (__bf16*)(ws + off); const int eW1n = 32 * 2  * 512; off += (size_t)eW1n * 2;
  __bf16* eW2s = (__bf16*)(ws + off); const int eW2n = 32 * 16 * 512; off += (size_t)eW2n * 2;
  __bf16* eW3s = (__bf16*)(ws + off); const int eW3n = 32 * 16 * 512; off += (size_t)eW3n * 2;
  __bf16* nW1s = (__bf16*)(ws + off); const int nW1n = 32 * 18 * 512; off += (size_t)nW1n * 2;
  __bf16* nW2s = (__bf16*)(ws + off); const int nW2n = 32 * 16 * 512; off += (size_t)nW2n * 2;
  __bf16* nW3s = (__bf16*)(ws + off); const int nW3n = 2  * 16 * 512; off += (size_t)nW3n * 2;
  __bf16* aggb = (__bf16*)(ws + off); // 20480 * 512 bf16 = 20 MB

  // weight prep (all tiny; one-shot every call for determinism)
  swizzle_w_kernel<<<(eW1n + 255) / 256, 256, 0, stream>>>(eW1, eW1s,  64, 512,  2, eW1n);
  swizzle_w_kernel<<<(eW2n + 255) / 256, 256, 0, stream>>>(eW2, eW2s, 512, 512, 16, eW2n);
  swizzle_w_kernel<<<(eW3n + 255) / 256, 256, 0, stream>>>(eW3, eW3s, 512, 512, 16, eW3n);
  swizzle_w_kernel<<<(nW1n + 255) / 256, 256, 0, stream>>>(nW1, nW1s, 548, 512, 18, nW1n);
  swizzle_w_kernel<<<(nW2n + 255) / 256, 256, 0, stream>>>(nW2, nW2s, 512, 512, 16, nW2n);
  swizzle_w_kernel<<<(nW3n + 255) / 256, 256, 0, stream>>>(nW3, nW3s, 512,  32, 16, nW3n);

  const size_t S1 = 12288 + 2 * 98304 + 20480 + 4 * 96 * sizeof(float) + 1280;   // 232192 B
  const size_t S2 = 73728 + 2 * 65536 + 4 * 64 * sizeof(float) + 65536;          // 271360 B

  edge_mlp_agg_kernel<<<2048, 256, S1, stream>>>(
      states, eW1s, eb1, eW2s, eb2, eg, ebt, eW3s, eb3, aggb);

  node_mlp_kernel<<<320, 256, S2, stream>>>(
      states, action, aggb, nW1s, nb1, nW2s, nb2, ng, nbt, nW3s, nb3, (float*)d_out);
}
